// graph_conv_v2_30193620091001
// MI455X (gfx1250) — matrix operand order repaired and checked by reading, not compiled, not run on hardware
//
#include <hip/hip_runtime.h>

typedef __bf16 bf16;
typedef __bf16 v16bf __attribute__((ext_vector_type(16)));
typedef __bf16 v8bf  __attribute__((ext_vector_type(8)));
typedef float  v8f   __attribute__((ext_vector_type(8)));

#define NB    8
#define NNODE 4096
#define NE    32768
#define FNODE 256
#define FEDGE 128
#define FIN   512
#define FCAT  640   // 2*FNODE + FEDGE

// ---- workspace layout (bytes) ----
// node_sums f32 : NB*NNODE*FNODE            = 33,554,432 B
// cnt       f32 : NB*NNODE                  =    131,072 B
// Wt_in   bf16  : FIN*FCAT  (N-major, [512][640])
// Wt_out  bf16  : FNODE*FIN ([256][512])
// Wt_edge bf16  : FEDGE*FNODE ([128][256])
#define WS_SUMS_OFF  0
#define WS_CNT_OFF   (33554432)
#define WS_WIN_OFF   (33554432 + 131072)
#define WS_WOUT_OFF  (WS_WIN_OFF  + FIN*FCAT*2)
#define WS_WEDGE_OFF (WS_WOUT_OFF + FNODE*FIN*2)

// output layout (float elements)
#define OUT_EDGES_OFF ((size_t)NB*NNODE*FNODE)                       // 8,388,608
#define OUT_SEND_OFF  (OUT_EDGES_OFF + (size_t)NB*NE*FEDGE)          // +33,554,432
#define OUT_RECV_OFF  (OUT_SEND_OFF + (size_t)NB*NE)

// ---------------- helper fragment loads ----------------
__device__ __forceinline__ v16bf load_a_lds(const bf16* base, int strideElems,
                                            int kt, int lane) {
  int m = lane & 15, g = lane >> 4;
  const bf16* p = base + m * strideElems + kt * 32 + g * 8;
  v8bf lo = *(const v8bf*)p;
  v8bf hi = *(const v8bf*)(p + 16);
  return __builtin_shufflevector(lo, hi, 0,1,2,3,4,5,6,7,8,9,10,11,12,13,14,15);
}

__device__ __forceinline__ v16bf load_b_w(const bf16* W, int Kdim,
                                          int nt, int kt, int lane) {
  int n = lane & 15, g = lane >> 4;
  const bf16* p = W + (size_t)(nt * 16 + n) * Kdim + kt * 32 + g * 8;
  v8bf lo = *(const v8bf*)p;
  v8bf hi = *(const v8bf*)(p + 16);
  return __builtin_shufflevector(lo, hi, 0,1,2,3,4,5,6,7,8,9,10,11,12,13,14,15);
}

// ---------------- kernel 0: zero sums + counts ----------------
__global__ void gnn_zero_kernel(float* sums, float* cnt) {
  size_t total = (size_t)NB * NNODE * FNODE;
  size_t ctot  = (size_t)NB * NNODE;
  for (size_t i = blockIdx.x * 256ull + threadIdx.x; i < total;
       i += (size_t)gridDim.x * 256ull)
    sums[i] = 0.0f;
  for (size_t i = blockIdx.x * 256ull + threadIdx.x; i < ctot;
       i += (size_t)gridDim.x * 256ull)
    cnt[i] = 0.0f;
}

// ---------------- kernel 1: convert + transpose weights to bf16 ----------------
__global__ void gnn_cvt_weights_kernel(const float* W_in, const float* W_out,
                                       const float* W_edge,
                                       bf16* WtIn, bf16* WtOut, bf16* WtEdge) {
  const int n_in = FIN * FCAT;      // 327,680
  const int n_out = FNODE * FIN;    // 131,072
  const int n_edge = FEDGE * FNODE; //  32,768
  for (int idx = blockIdx.x * 256 + threadIdx.x; idx < n_in + n_out + n_edge;
       idx += gridDim.x * 256) {
    if (idx < n_in) {
      int n = idx / FCAT, k = idx % FCAT;
      WtIn[idx] = (bf16)W_in[(size_t)k * FIN + n];
    } else if (idx < n_in + n_out) {
      int i = idx - n_in;
      int n = i / FIN, k = i % FIN;
      WtOut[i] = (bf16)W_out[(size_t)k * FNODE + n];
    } else {
      int i = idx - n_in - n_out;
      int n = i / FNODE, k = i % FNODE;
      WtEdge[i] = (bf16)W_edge[(size_t)k * FEDGE + n];
    }
  }
}

// ---------------- kernel 2: fused edge MLP ----------------
// block: 256 threads (8 waves), tile = 16 edges. grid = (NE/16, NB)
#define D_STRIDE  648   // 640 + 8 pad (bf16 elems), 1296 B rows (16B aligned)
#define H_STRIDE  520   // 512 + 8
#define E2_STRIDE 264   // 256 + 8

__global__ __launch_bounds__(256)
void gnn_edge_mlp_kernel(const float* __restrict__ nodes,
                         const float* __restrict__ edges,
                         const int* __restrict__ senders,
                         const int* __restrict__ receivers,
                         const bf16* __restrict__ WtIn,
                         const bf16* __restrict__ WtOut,
                         const bf16* __restrict__ WtEdge,
                         const float* __restrict__ b_in,
                         const float* __restrict__ b_out,
                         const float* __restrict__ b_edge,
                         float* __restrict__ out,       // full d_out
                         float* __restrict__ nodeSums,
                         float* __restrict__ cnt) {
  // region A: d tile (16x648 bf16 = 20736B) then e2_f32 (16384B) + e2_bf16 (8448B)
  __shared__ __align__(32) unsigned char smemA[24832];
  // region B: h tile (16x520 bf16 = 16640B)
  __shared__ __align__(32) unsigned char smemB[16640];
  __shared__ int rIdx[16];
  __shared__ int sIdx[16];

  bf16*  dT  = (bf16*)smemA;
  float* e2f = (float*)smemA;                 // 16 x 256 f32
  bf16*  e2b = (bf16*)(smemA + 16384);        // 16 x 264 bf16
  bf16*  hT  = (bf16*)smemB;                  // 16 x 520 bf16

  const int tid  = threadIdx.x;
  const int lane = tid & 31;
  const int wave = tid >> 5;
  const int b    = blockIdx.y;
  const int e0   = blockIdx.x * 16;

  if (tid < 16) {
    rIdx[tid] = receivers[(size_t)b * NE + e0 + tid];
    sIdx[tid] = senders[(size_t)b * NE + e0 + tid];
  }
  __syncthreads();

  // ---- phase 1: gather [recv || edge || send] -> bf16 LDS tile ----
  for (int it = tid; it < 16 * 160; it += 256) {
    int row = it / 160;
    int col = (it % 160) * 4;
    const float* src;
    if (col < FNODE)
      src = nodes + ((size_t)b * NNODE + rIdx[row]) * FNODE + col;
    else if (col < FNODE + FEDGE)
      src = edges + ((size_t)b * NE + e0 + row) * FEDGE + (col - FNODE);
    else
      src = nodes + ((size_t)b * NNODE + sIdx[row]) * FNODE + (col - FNODE - FEDGE);
    float4 v = *(const float4*)src;
    bf16* dst = dT + row * D_STRIDE + col;
    dst[0] = (bf16)v.x; dst[1] = (bf16)v.y; dst[2] = (bf16)v.z; dst[3] = (bf16)v.w;
  }
  __syncthreads();

  const int g8   = (lane >> 4) * 8;
  const int nlo  = lane & 15;

  // ---- phase 2: GEMM1  d[16,640] @ W_in -> h[16,512], relu ----
  {
    v8f acc[4] = {};
    for (int kt = 0; kt < FCAT / 32; ++kt) {
      v16bf a = load_a_lds(dT, D_STRIDE, kt, lane);
#pragma unroll
      for (int j = 0; j < 4; ++j) {
        int nt = wave + 8 * j;
        v16bf w = load_b_w(WtIn, FCAT, nt, kt, lane);
        acc[j] = __builtin_amdgcn_wmma_f32_16x16x32_bf16(
            false, a, false, w, (short)0, acc[j], false, false);
      }
    }
#pragma unroll
    for (int j = 0; j < 4; ++j) {
      int colN = (wave + 8 * j) * 16 + nlo;
      float bias = b_in[colN];
#pragma unroll
      for (int v = 0; v < 8; ++v) {
        float x = acc[j][v] + bias;
        x = x > 0.0f ? x : 0.0f;
        hT[(v + g8) * H_STRIDE + colN] = (bf16)x;
      }
    }
  }
  __syncthreads();

  // ---- phase 3: GEMM2  h[16,512] @ W_out -> e2[16,256], relu ----
  {
    v8f acc[2] = {};
    for (int kt = 0; kt < FIN / 32; ++kt) {
      v16bf a = load_a_lds(hT, H_STRIDE, kt, lane);
#pragma unroll
      for (int j = 0; j < 2; ++j) {
        int nt = wave + 8 * j;
        v16bf w = load_b_w(WtOut, FIN, nt, kt, lane);
        acc[j] = __builtin_amdgcn_wmma_f32_16x16x32_bf16(
            false, a, false, w, (short)0, acc[j], false, false);
      }
    }
    __syncthreads();   // all reads of dT-region done; safe to overwrite with e2
#pragma unroll
    for (int j = 0; j < 2; ++j) {
      int colN = (wave + 8 * j) * 16 + nlo;
      float bias = b_out[colN];
#pragma unroll
      for (int v = 0; v < 8; ++v) {
        float x = acc[j][v] + bias;
        x = x > 0.0f ? x : 0.0f;
        int row = v + g8;
        e2f[row * FNODE + colN] = x;
        e2b[row * E2_STRIDE + colN] = (bf16)x;
      }
    }
  }
  __syncthreads();

  // ---- phase 4: GEMM3  e2[16,256] @ W_edge -> edges_new[16,128], relu ----
  {
    v8f acc = {};
    int nt = wave;  // 8 waves cover 8 N-tiles of 128
    for (int kt = 0; kt < FNODE / 32; ++kt) {
      v16bf a = load_a_lds(e2b, E2_STRIDE, kt, lane);
      v16bf w = load_b_w(WtEdge, FNODE, nt, kt, lane);
      acc = __builtin_amdgcn_wmma_f32_16x16x32_bf16(
          false, a, false, w, (short)0, acc, false, false);
    }
    int colN = nt * 16 + nlo;
    float bias = b_edge[colN];
    float* oe = out + OUT_EDGES_OFF;
#pragma unroll
    for (int v = 0; v < 8; ++v) {
      float x = acc[v] + bias;
      x = x > 0.0f ? x : 0.0f;
      oe[((size_t)b * NE + e0 + (v + g8)) * FEDGE + colN] = x;
    }
  }

  // ---- phase 5: segment-sum of e2 over receivers (hw f32 atomics) ----
  for (int it = tid; it < 16 * FNODE; it += 256) {
    int row = it >> 8, col = it & 255;
    unsafeAtomicAdd(nodeSums + ((size_t)b * NNODE + rIdx[row]) * FNODE + col,
                    e2f[row * FNODE + col]);
  }
  if (tid < 16)
    unsafeAtomicAdd(cnt + (size_t)b * NNODE + rIdx[tid], 1.0f);
}

// ---------------- kernel 3: mean finalize ----------------
__global__ void gnn_finalize_kernel(const float* sums, const float* cnt,
                                    float* out) {
  size_t total = (size_t)NB * NNODE * FNODE;
  for (size_t i = blockIdx.x * 256ull + threadIdx.x; i < total;
       i += (size_t)gridDim.x * 256ull) {
    float c = cnt[i >> 8];
    out[i] = sums[i] / fmaxf(c, 1.0f);
  }
}

// ---------------- kernel 4: pass-through senders/receivers (bit copy) ----------
__global__ void gnn_copy_idx_kernel(const int* senders, const int* receivers,
                                    int* outAsInt) {
  const int n = NB * NE;
  for (int i = blockIdx.x * 256 + threadIdx.x; i < 2 * n; i += gridDim.x * 256) {
    if (i < n) outAsInt[OUT_SEND_OFF + i] = senders[i];
    else       outAsInt[OUT_SEND_OFF + i] = receivers[i - n];  // contiguous tail
  }
}

extern "C" void kernel_launch(void* const* d_in, const int* in_sizes, int n_in,
                              void* d_out, int out_size, void* d_ws, size_t ws_size,
                              hipStream_t stream) {
  const float* nodes     = (const float*)d_in[0];
  const float* edges     = (const float*)d_in[1];
  const int*   senders   = (const int*)d_in[2];
  const int*   receivers = (const int*)d_in[3];
  const float* W_in      = (const float*)d_in[4];
  const float* b_in      = (const float*)d_in[5];
  const float* W_out     = (const float*)d_in[6];
  const float* b_out     = (const float*)d_in[7];
  const float* W_edge    = (const float*)d_in[8];
  const float* b_edge    = (const float*)d_in[9];

  unsigned char* ws = (unsigned char*)d_ws;
  float* nodeSums = (float*)(ws + WS_SUMS_OFF);
  float* cnt      = (float*)(ws + WS_CNT_OFF);
  bf16*  WtIn     = (bf16*)(ws + WS_WIN_OFF);
  bf16*  WtOut    = (bf16*)(ws + WS_WOUT_OFF);
  bf16*  WtEdge   = (bf16*)(ws + WS_WEDGE_OFF);
  float* out      = (float*)d_out;

  gnn_zero_kernel<<<4096, 256, 0, stream>>>(nodeSums, cnt);
  gnn_cvt_weights_kernel<<<1920, 256, 0, stream>>>(W_in, W_out, W_edge,
                                                   WtIn, WtOut, WtEdge);
  dim3 grid(NE / 16, NB);
  gnn_edge_mlp_kernel<<<grid, 256, 0, stream>>>(
      nodes, edges, senders, receivers, WtIn, WtOut, WtEdge,
      b_in, b_out, b_edge, out, nodeSums, cnt);
  gnn_finalize_kernel<<<8192, 256, 0, stream>>>(nodeSums, cnt, out);
  gnn_copy_idx_kernel<<<2048, 256, 0, stream>>>(senders, receivers, (int*)d_out);
}
